// Model_67362267070927
// MI455X (gfx1250) — matrix operand order repaired and checked by reading, not compiled, not run on hardware
//
#include <hip/hip_runtime.h>

// ---------------------------------------------------------------------------
// CDNA5 (gfx1250) hetero-GraphSAGE inference.
// GEMMs: v_wmma_f32_16x16x32_bf16 (wave32). Weights are pre-converted to bf16
// once (~1.7MB) so B fragments load directly (2x b128, no packing); A stays
// f32 (activations) and is packed in-register (8 pack ops / k-step). Each wave
// computes a 16x64 C strip (4 accumulators sharing one A fragment).
// Segment-mean division is folded into the GEMM epilogue; degree counting is
// folded into the scatter pass. All M%16==0, N%64==0, K%32==0 -> no tails,
// EXEC all-ones for WMMA (ISA 7.12).
// ---------------------------------------------------------------------------

#define HDIM 256
#define WPB  4   // waves per block in GEMM kernels

typedef __attribute__((ext_vector_type(16))) __bf16        v16bf;
typedef __attribute__((ext_vector_type(8)))  float         v8f;
typedef __attribute__((ext_vector_type(8)))  unsigned int  v8u;

struct F4x4 { float4 a, b, c, d; };

// pack two f32 into one dword of two bf16 (truncation)
__device__ __forceinline__ unsigned pack_bf2(float f0, float f1) {
    unsigned u0 = __builtin_bit_cast(unsigned, f0);
    unsigned u1 = __builtin_bit_cast(unsigned, f1);
    return (u0 >> 16) | (u1 & 0xFFFF0000u);
}

// A fragment staging: floats p[0..15]
__device__ __forceinline__ F4x4 loadA16(const float* __restrict__ p) {
    const float4* q = reinterpret_cast<const float4*>(p);
    const float4* r = reinterpret_cast<const float4*>(p + 8);
    F4x4 v; v.a = q[0]; v.b = q[1]; v.c = r[0]; v.d = r[1];
    return v;
}

__device__ __forceinline__ v16bf pack16(const F4x4& f) {
    v8u u;
    u[0] = pack_bf2(f.a.x, f.a.y); u[1] = pack_bf2(f.a.z, f.a.w);
    u[2] = pack_bf2(f.b.x, f.b.y); u[3] = pack_bf2(f.b.z, f.b.w);
    u[4] = pack_bf2(f.c.x, f.c.y); u[5] = pack_bf2(f.c.z, f.c.w);
    u[6] = pack_bf2(f.d.x, f.d.y); u[7] = pack_bf2(f.d.z, f.d.w);
    return __builtin_bit_cast(v16bf, u);
}

// B fragment: 16 contiguous bf16 at p (32 bytes, 2x b128)
__device__ __forceinline__ v16bf loadBf(const unsigned short* __restrict__ p) {
    const uint4* q = reinterpret_cast<const uint4*>(p);
    uint4 x = q[0], y = q[1];
    v8u u;
    u[0] = x.x; u[1] = x.y; u[2] = x.z; u[3] = x.w;
    u[4] = y.x; u[5] = y.y; u[6] = y.z; u[7] = y.w;
    return __builtin_bit_cast(v16bf, u);
}

// epilogue for one 16x16 tile held in a v8f accumulator
// rowdiv: if non-null, scale row r by 1/max(rowdiv[r],1)  (segment mean)
__device__ __forceinline__ void store_tile(v8f acc, float* __restrict__ C,
                                           const float* __restrict__ bias,
                                           const float* __restrict__ add,
                                           const float* __restrict__ rowdiv,
                                           int mrow, int col, int N, int half, int relu) {
    const float bb = bias ? bias[col] : 0.0f;
#pragma unroll
    for (int i = 0; i < 8; ++i) {
        const int row = mrow + i + half * 8;   // C/D layout: VGPR i -> M=i (+8 upper half)
        const size_t o = (size_t)row * N + col;
        float v = acc[i];
        if (rowdiv) v *= 1.0f / fmaxf(rowdiv[row], 1.0f);
        v += bb;
        if (add)  v += add[o];
        if (relu) v = fmaxf(v, 0.0f);
        C[o] = v;
    }
}

// C[M,N] = rowscale? * (A[M,K] @ Wbf[N,K]^T) (+bias) (+add) (relu)
// M%16==0, N%64==0, K%32==0. One wave -> 16 rows x 64 cols.
__global__ void __launch_bounds__(WPB * 32, 1)
wmma_gemm(const float* __restrict__ A, const unsigned short* __restrict__ W,
          const float* __restrict__ bias, const float* __restrict__ add,
          const float* __restrict__ rowdiv,
          float* __restrict__ C, int M, int N, int K, int relu) {
    const int lane = threadIdx.x & 31;
    const int half = lane >> 4;
    const int r    = lane & 15;
    const int nstrips = N >> 6;               // strips of 64 columns
    const int tiles   = (M >> 4) * nstrips;
    const int gw = blockIdx.x * WPB + (threadIdx.x >> 5);
    if (gw >= tiles) return;                  // wave-uniform exit
    const int mrow = (gw / nstrips) << 4;
    const int ncol = (gw % nstrips) << 6;

    const float*          arow  = A + (size_t)(mrow + r) * K;
    const unsigned short* brow0 = W + (size_t)(ncol +  0 + r) * K;
    const unsigned short* brow1 = W + (size_t)(ncol + 16 + r) * K;
    const unsigned short* brow2 = W + (size_t)(ncol + 32 + r) * K;
    const unsigned short* brow3 = W + (size_t)(ncol + 48 + r) * K;

    v8f acc0 = {}, acc1 = {}, acc2 = {}, acc3 = {};
    for (int k0 = 0; k0 < K; k0 += 32) {
        const int kb = k0 + half * 16;
        F4x4  sa  = loadA16(arow + k0 + half * 16);
        v16bf bf0 = loadBf(brow0 + kb);
        v16bf bf1 = loadBf(brow1 + kb);
        v16bf bf2 = loadBf(brow2 + kb);
        v16bf bf3 = loadBf(brow3 + kb);
        v16bf af  = pack16(sa);
        acc0 = __builtin_amdgcn_wmma_f32_16x16x32_bf16(false, af, false, bf0, (short)0, acc0, false, false);
        acc1 = __builtin_amdgcn_wmma_f32_16x16x32_bf16(false, af, false, bf1, (short)0, acc1, false, false);
        acc2 = __builtin_amdgcn_wmma_f32_16x16x32_bf16(false, af, false, bf2, (short)0, acc2, false, false);
        acc3 = __builtin_amdgcn_wmma_f32_16x16x32_bf16(false, af, false, bf3, (short)0, acc3, false, false);
    }
    store_tile(acc0, C, bias, add, rowdiv, mrow, ncol +  0 + r, N, half, relu);
    store_tile(acc1, C, bias, add, rowdiv, mrow, ncol + 16 + r, N, half, relu);
    store_tile(acc2, C, bias, add, rowdiv, mrow, ncol + 32 + r, N, half, relu);
    store_tile(acc3, C, bias, add, rowdiv, mrow, ncol + 48 + r, N, half, relu);
}

// classifier GEMM: A row m = concat(HP[idxp[m]], HS[idxs[m]]) (K=512), relu out
__global__ void __launch_bounds__(WPB * 32, 1)
wmma_gemm_pair(const float* __restrict__ HP, const float* __restrict__ HS,
               const int* __restrict__ idxp, const int* __restrict__ idxs,
               const unsigned short* __restrict__ W, const float* __restrict__ bias,
               float* __restrict__ C, int M, int N) {
    const int K = 512;
    const int lane = threadIdx.x & 31;
    const int half = lane >> 4;
    const int r    = lane & 15;
    const int nstrips = N >> 6;
    const int tiles   = (M >> 4) * nstrips;
    const int gw = blockIdx.x * WPB + (threadIdx.x >> 5);
    if (gw >= tiles) return;
    const int mrow = (gw / nstrips) << 4;
    const int ncol = (gw % nstrips) << 6;

    const float* arowP = HP + (size_t)idxp[mrow + r] * HDIM;
    const float* arowS = HS + (size_t)idxs[mrow + r] * HDIM;
    const unsigned short* brow0 = W + (size_t)(ncol +  0 + r) * K;
    const unsigned short* brow1 = W + (size_t)(ncol + 16 + r) * K;
    const unsigned short* brow2 = W + (size_t)(ncol + 32 + r) * K;
    const unsigned short* brow3 = W + (size_t)(ncol + 48 + r) * K;

    v8f acc0 = {}, acc1 = {}, acc2 = {}, acc3 = {};
    for (int k0 = 0; k0 < K; k0 += 32) {       // 256 | 32 -> never straddles the concat
        const float* arow = (k0 < 256) ? arowP : arowS;
        const int kk = (k0 < 256) ? k0 : (k0 - 256);
        const int kb = k0 + half * 16;
        F4x4  sa  = loadA16(arow + kk + half * 16);
        v16bf bf0 = loadBf(brow0 + kb);
        v16bf bf1 = loadBf(brow1 + kb);
        v16bf bf2 = loadBf(brow2 + kb);
        v16bf bf3 = loadBf(brow3 + kb);
        v16bf af  = pack16(sa);
        acc0 = __builtin_amdgcn_wmma_f32_16x16x32_bf16(false, af, false, bf0, (short)0, acc0, false, false);
        acc1 = __builtin_amdgcn_wmma_f32_16x16x32_bf16(false, af, false, bf1, (short)0, acc1, false, false);
        acc2 = __builtin_amdgcn_wmma_f32_16x16x32_bf16(false, af, false, bf2, (short)0, acc2, false, false);
        acc3 = __builtin_amdgcn_wmma_f32_16x16x32_bf16(false, af, false, bf3, (short)0, acc3, false, false);
    }
    store_tile(acc0, C, bias, nullptr, nullptr, mrow, ncol +  0 + r, N, half, 1);
    store_tile(acc1, C, bias, nullptr, nullptr, mrow, ncol + 16 + r, N, half, 1);
    store_tile(acc2, C, bias, nullptr, nullptr, mrow, ncol + 32 + r, N, half, 1);
    store_tile(acc3, C, bias, nullptr, nullptr, mrow, ncol + 48 + r, N, half, 1);
}

// f32 -> packed bf16 pairs (weights; tiny)
__global__ void cvt_bf16(unsigned* __restrict__ dst, const float* __restrict__ src, int npairs) {
    int t = blockIdx.x * blockDim.x + threadIdx.x;
    if (t < npairs) dst[t] = pack_bf2(src[2 * t], src[2 * t + 1]);
}

__global__ void zero_k(float4* __restrict__ p, long n4) {
    long t = (long)blockIdx.x * blockDim.x + threadIdx.x;
    if (t < n4) p[t] = make_float4(0.f, 0.f, 0.f, 0.f);
}

// dst[m, :] += emb[idx[m], :]
__global__ void add_gather(float* __restrict__ dst, const float* __restrict__ emb,
                           const int* __restrict__ idx, int rows) {
    long t = (long)blockIdx.x * blockDim.x + threadIdx.x;
    if (t >= (long)rows * (HDIM / 4)) return;
    const int row = (int)(t >> 6);
    const int f   = ((int)t & 63) << 2;
    const float4 e = *reinterpret_cast<const float4*>(emb + (size_t)idx[row] * HDIM + f);
    float4* p = reinterpret_cast<float4*>(dst + (size_t)row * HDIM + f);
    float4 v = *p;
    v.x += e.x; v.y += e.y; v.z += e.z; v.w += e.w;
    *p = v;
}

// dst[sidx[e], :] += src[gidx[e], :]  and  cnt[sidx[e]] += 1
// (64 threads/edge, float4 each; sub-thread 0 also counts)
__global__ void scatter_add(const float* __restrict__ src, const int* __restrict__ gidx,
                            const int* __restrict__ sidx, float* __restrict__ dst,
                            float* __restrict__ cnt, int E) {
    long t = (long)blockIdx.x * blockDim.x + threadIdx.x;
    const int e = (int)(t >> 6);
    if (e >= E) return;
    const int sub = (int)t & 63;
    const int f = sub << 2;
    const int s = sidx[e];
    const float4 v = *reinterpret_cast<const float4*>(src + (size_t)gidx[e] * HDIM + f);
    float* d = dst + (size_t)s * HDIM + f;
    atomicAdd(d + 0, v.x); atomicAdd(d + 1, v.y);
    atomicAdd(d + 2, v.z); atomicAdd(d + 3, v.w);
    if (sub == 0) atomicAdd(&cnt[s], 1.0f);
}

// out[l] = dot(Hm[l,:], w) + b[0] ; one wave per row, shfl_xor butterfly (wave32)
__global__ void out_dot(const float* __restrict__ Hm, const float* __restrict__ w,
                        const float* __restrict__ b, float* __restrict__ out, int L) {
    const int wid  = (int)(((long)blockIdx.x * blockDim.x + threadIdx.x) >> 5);
    const int lane = threadIdx.x & 31;
    if (wid >= L) return;
    const float* row = Hm + (size_t)wid * HDIM;
    float s = 0.0f;
#pragma unroll
    for (int i = 0; i < HDIM / 32; ++i) s += row[lane + i * 32] * w[lane + i * 32];
    for (int off = 16; off; off >>= 1) s += __shfl_xor(s, off, 32);
    if (lane == 0) out[wid] = s + b[0];
}

// ---------------------------------------------------------------------------

static inline void launch_gemm(const float* A, const unsigned short* W, const float* bias,
                               const float* add, const float* rowdiv, float* C,
                               int M, int N, int K, int relu, hipStream_t st) {
    int tiles = (M / 16) * (N / 64);
    int blocks = (tiles + WPB - 1) / WPB;
    wmma_gemm<<<blocks, WPB * 32, 0, st>>>(A, W, bias, add, rowdiv, C, M, N, K, relu);
}

extern "C" void kernel_launch(void* const* d_in, const int* in_sizes, int n_in,
                              void* d_out, int out_size, void* d_ws, size_t ws_size,
                              hipStream_t stream) {
    const float* x_paper  = (const float*)d_in[0];
    const float* x_sw     = (const float*)d_in[1];
    const float* pemb     = (const float*)d_in[2];
    const float* semb     = (const float*)d_in[3];
    const float* bp       = (const float*)d_in[5];
    const float* bs       = (const float*)d_in[7];
    const float* l1m_bl = (const float*)d_in[9];
    const float* l1r_bl = (const float*)d_in[12];
    const float* l2m_bl = (const float*)d_in[15];
    const float* l2r_bl = (const float*)d_in[18];
    const float* c1_b = (const float*)d_in[21];
    const float* c2_W = (const float*)d_in[22];   const float* c2_b = (const float*)d_in[23];
    const int* paper_id = (const int*)d_in[24];
    const int* sw_id    = (const int*)d_in[25];
    const int* edge_src = (const int*)d_in[26];
    const int* edge_dst = (const int*)d_in[27];
    const int* label_p  = (const int*)d_in[28];
    const int* label_s  = (const int*)d_in[29];
    float* out = (float*)d_out;

    const int NP  = in_sizes[24];
    const int NS  = in_sizes[25];
    const int E   = in_sizes[26];
    const int L   = in_sizes[28];
    const int EMB = in_sizes[0] / NP;
    const int Hh  = in_sizes[5];          // 256

    // workspace partition (floats)
    float* w0 = (float*)d_ws;
    size_t o = 0;
    float* hpA  = w0 + o; o += (size_t)NP * Hh;
    float* hpB  = w0 + o; o += (size_t)NP * Hh;
    float* aggP = w0 + o; o += (size_t)NP * Hh;   // reused as classifier hidden (L == NP)
    float* tP   = w0 + o; o += (size_t)NP * Hh;
    float* hsA  = w0 + o; o += (size_t)NS * Hh;
    float* hsB  = w0 + o; o += (size_t)NS * Hh;
    float* aggS = w0 + o; o += (size_t)NS * Hh;
    float* tS   = w0 + o; o += (size_t)NS * Hh;
    float* cntP = w0 + o; o += (size_t)NP;
    float* cntS = w0 + o; o += (size_t)NS;
    (void)ws_size; (void)n_in; (void)out_size;

    const int TPB = 256;
    auto gr = [](long n, int b) { return (int)((n + b - 1) / b); };

    // ---- pre-convert all weight matrices to bf16 (one-time, ~1.7MB) ----
    // order: Wp(4), Ws(6), l1m_Wl(8), l1m_Wr(10), l1r_Wl(11), l1r_Wr(13),
    //        l2m_Wl(14), l2m_Wr(16), l2r_Wl(17), l2r_Wr(19), c1_W(20)
    const int widx[11] = {4, 6, 8, 10, 11, 13, 14, 16, 17, 19, 20};
    unsigned short* wbf[11];
    unsigned short* bbase = (unsigned short*)(w0 + o);
    size_t bo = 0;
    for (int i = 0; i < 11; ++i) {
        wbf[i] = bbase + bo;
        bo += (size_t)in_sizes[widx[i]];
        const int npairs = in_sizes[widx[i]] / 2;
        cvt_bf16<<<gr(npairs, TPB), TPB, 0, stream>>>((unsigned*)wbf[i],
                                                      (const float*)d_in[widx[i]], npairs);
    }
    const unsigned short* bWp = wbf[0];   const unsigned short* bWs = wbf[1];
    const unsigned short* Wl_m[2] = {wbf[2], wbf[6]};
    const unsigned short* Wr_m[2] = {wbf[3], wbf[7]};
    const unsigned short* Wl_r[2] = {wbf[4], wbf[8]};
    const unsigned short* Wr_r[2] = {wbf[5], wbf[9]};
    const unsigned short* bc1 = wbf[10];
    const float* bl_m[2] = {l1m_bl, l2m_bl};
    const float* bl_r[2] = {l1r_bl, l2r_bl};

    // ---- input projections + id-embedding adds ----
    launch_gemm(x_paper, bWp, bp, nullptr, nullptr, hpA, NP, Hh, EMB, 0, stream);
    add_gather<<<gr((long)NP * 64, TPB), TPB, 0, stream>>>(hpA, pemb, paper_id, NP);
    launch_gemm(x_sw, bWs, bs, nullptr, nullptr, hsA, NS, Hh, EMB, 0, stream);
    add_gather<<<gr((long)NS * 64, TPB), TPB, 0, stream>>>(hsA, semb, sw_id, NS);

    const float* inP = hpA; const float* inS = hsA;
    float* outP = hpB;      float* outS = hsB;

    for (int lyr = 0; lyr < 2; ++lyr) {
        const int relu = (lyr == 0) ? 1 : 0;
        // paper -> software aggregation (mean over edge_dst segments; /cnt folded in GEMM)
        zero_k<<<gr((long)NS * Hh / 4, TPB), TPB, 0, stream>>>((float4*)aggS, (long)NS * Hh / 4);
        zero_k<<<gr((NS + 3) / 4, TPB), TPB, 0, stream>>>((float4*)cntS, (NS + 3) / 4);
        scatter_add<<<gr((long)E * 64, TPB), TPB, 0, stream>>>(inP, edge_src, edge_dst, aggS, cntS, E);
        launch_gemm(aggS, Wl_m[lyr], bl_m[lyr], nullptr, cntS, tS, NS, Hh, Hh, 0, stream);
        launch_gemm(inS, Wr_m[lyr], nullptr, tS, nullptr, outS, NS, Hh, Hh, relu, stream);

        // software -> paper aggregation (mean over edge_src segments)
        zero_k<<<gr((long)NP * Hh / 4, TPB), TPB, 0, stream>>>((float4*)aggP, (long)NP * Hh / 4);
        zero_k<<<gr((NP + 3) / 4, TPB), TPB, 0, stream>>>((float4*)cntP, (NP + 3) / 4);
        scatter_add<<<gr((long)E * 64, TPB), TPB, 0, stream>>>(inS, edge_dst, edge_src, aggP, cntP, E);
        launch_gemm(aggP, Wl_r[lyr], bl_r[lyr], nullptr, cntP, tP, NP, Hh, Hh, 0, stream);
        launch_gemm(inP, Wr_r[lyr], nullptr, tP, nullptr, outP, NP, Hh, Hh, relu, stream);

        // ping-pong
        const float* nInP = outP; const float* nInS = outS;
        outP = (float*)inP; outS = (float*)inS;
        inP = nInP; inS = nInS;
    }

    // ---- link classifier: h = relu(concat(hp2[lp], hs2[ls]) @ c1_W^T + c1_b) ----
    float* hMid = aggP;   // free after layers; L*H == NP*H
    {
        int tiles = (L / 16) * (Hh / 64);
        int blocks = (tiles + WPB - 1) / WPB;
        wmma_gemm_pair<<<blocks, WPB * 32, 0, stream>>>(inP, inS, label_p, label_s,
                                                        bc1, c1_b, hMid, L, Hh);
    }
    // ---- out = h @ c2_W^T + c2_b ----
    out_dot<<<gr((long)L * 32, TPB), TPB, 0, stream>>>(hMid, c2_W, c2_b, out, L);
}
